// DistributionalQNetwork_4741643894997
// MI455X (gfx1250) — matrix operand order repaired and checked by reading, compiled, not run on hardware
//
#include <hip/hip_runtime.h>
#include <hip/hip_bf16.h>

typedef __attribute__((ext_vector_type(16))) _Float16 v16h;
typedef __attribute__((ext_vector_type(8)))  float    v8f;

#define OBS_DIM 96
#define ACT_DIM 32
#define IN_DIM  128
#define H_DIM   512
#define NATOMS  251
#define NPAD    256
#define MT      64      // rows per block

// LDS strides (padded: row stride mod 64 banks == 4 -> conflict-free b128 loads)
#define SA  136         // x tile stride, halfs   (64 x 136 x 2  = 17408 B)
#define SH  520         // h tile stride, halfs   (64 x 520 x 2  = 66560 B)
#define SL  260         // logits stride, floats  (64 x 260 x 4  = 66560 B)
#define SP  257         // proj stride, floats    (64 x 257 x 4  = 65792 B)

#define OFF_A    0
#define OFF_H1   17408
#define OFF_H2   83968
#define OFF_LOG  0        // aliases sA+sH1 (dead after layer 2 reads)
#define OFF_PROJ 83968    // aliases sH2   (dead after layer 3 reads)
#define SMEM_BYTES 150528

// ---------------- weight prep: f32 -> f16, transpose to [N][K], pad W3 ----------------
__global__ void prep_weights(const float* __restrict__ W1, const float* __restrict__ W2,
                             const float* __restrict__ W3,
                             _Float16* __restrict__ W1t, _Float16* __restrict__ W2t,
                             _Float16* __restrict__ W3t) {
  int stride = gridDim.x * blockDim.x;
  int g = blockIdx.x * blockDim.x + threadIdx.x;
  for (int i = g; i < IN_DIM * H_DIM; i += stride) {       // W1t[h][k] = W1[k][h]
    int h = i / IN_DIM, k = i % IN_DIM;
    W1t[i] = (_Float16)W1[(size_t)k * H_DIM + h];
  }
  for (int i = g; i < H_DIM * H_DIM; i += stride) {        // W2t[h][k] = W2[k][h]
    int h = i / H_DIM, k = i % H_DIM;
    W2t[i] = (_Float16)W2[(size_t)k * H_DIM + h];
  }
  for (int i = g; i < NPAD * H_DIM; i += stride) {         // W3t[n][k], pad n>=251 with 0
    int n = i / H_DIM, k = i % H_DIM;
    W3t[i] = (n < NATOMS) ? (_Float16)W3[(size_t)k * NATOMS + n] : (_Float16)0.f;
  }
}

// A fragment (16x32 f16): lanes 0-15 -> row m, K {k0..k0+7, k0+16..k0+23}
//                         lanes16-31 -> row m, K {k0+8..k0+15, k0+24..k0+31}
__device__ __forceinline__ v16h load_a16(const _Float16* p0) {
  union { v16h v; uint4 q[2]; } u;
  u.q[0] = *(const uint4*)(p0);        // 8 halfs
  u.q[1] = *(const uint4*)(p0 + 16);   // 8 halfs, +16 half gap
  return u.v;
}
// B fragment (32x16 f16) from transposed weights [N][K]:
// lanes 0-15 -> col n, K {k0..k0+7, k0+16..k0+23}; lanes 16-31 -> K {k0+8..k0+15, k0+24..k0+31}
__device__ __forceinline__ v16h load_b16(const _Float16* p0) {
  union { v16h v; uint4 q[2]; } u;
  u.q[0] = *(const uint4*)(p0);
  u.q[1] = *(const uint4*)(p0 + 16);
  return u.v;
}

__global__ void __launch_bounds__(256)
fused_c51(const float* __restrict__ obs,  const float* __restrict__ act,
          const float* __restrict__ rew,  const float* __restrict__ boot,
          const float* __restrict__ disc, const float* __restrict__ qsup,
          const float* __restrict__ b1,   const float* __restrict__ b2,
          const float* __restrict__ b3,
          const _Float16* __restrict__ W1t, const _Float16* __restrict__ W2t,
          const _Float16* __restrict__ W3t, float* __restrict__ out) {
  extern __shared__ char smem[];
  _Float16* sA   = (_Float16*)(smem + OFF_A);
  _Float16* sH1  = (_Float16*)(smem + OFF_H1);
  _Float16* sH2  = (_Float16*)(smem + OFF_H2);
  float*    sLog = (float*)(smem + OFF_LOG);
  float*    sProj= (float*)(smem + OFF_PROJ);

  const int tid  = threadIdx.x;
  const int lane = tid & 31;
  const int wv   = tid >> 5;          // 8 waves
  const int m0   = blockIdx.x * MT;
  const int lrow = lane & 15;
  const int lhi  = lane >> 4;

  // ---- stage x = concat(obs, act) tile, f32 -> f16 ----
  for (int i = tid; i < MT * IN_DIM; i += 256) {
    int r = i >> 7, c = i & 127;
    float v = (c < OBS_DIM) ? obs[(size_t)(m0 + r) * OBS_DIM + c]
                            : act[(size_t)(m0 + r) * ACT_DIM + (c - OBS_DIM)];
    sA[r * SA + c] = (_Float16)v;
  }
  __syncthreads();

  // ---- layer 1: [64,128] @ [128,512] -> leaky -> sH1 ----
  {
    const int nb = wv * 64;
    v8f c[4][4] = {};
    for (int k0 = 0; k0 < IN_DIM; k0 += 32) {
      v16h af[4];
#pragma unroll
      for (int mi = 0; mi < 4; ++mi)
        af[mi] = load_a16(sA + (mi * 16 + lrow) * SA + k0 + lhi * 8);
#pragma unroll
      for (int ni = 0; ni < 4; ++ni) {
        v16h bf = load_b16(W1t + (size_t)(nb + ni * 16 + lrow) * IN_DIM + k0 + lhi * 8);
#pragma unroll
        for (int mi = 0; mi < 4; ++mi)
          c[mi][ni] = __builtin_amdgcn_wmma_f32_16x16x32_f16(
              false, af[mi], false, bf, (short)0, c[mi][ni], false, false);
      }
    }
#pragma unroll
    for (int mi = 0; mi < 4; ++mi)
#pragma unroll
      for (int ni = 0; ni < 4; ++ni)
#pragma unroll
        for (int j = 0; j < 8; ++j) {
          int r  = mi * 16 + lhi * 8 + j;
          int cc = nb + ni * 16 + lrow;
          float v = c[mi][ni][j] + b1[cc];
          v = v > 0.f ? v : 0.01f * v;
          sH1[r * SH + cc] = (_Float16)v;
        }
  }
  __syncthreads();

  // ---- layer 2: [64,512] @ [512,512] -> leaky -> sH2 ----
  {
    const int nb = wv * 64;
    v8f c[4][4] = {};
    for (int k0 = 0; k0 < H_DIM; k0 += 32) {
      v16h af[4];
#pragma unroll
      for (int mi = 0; mi < 4; ++mi)
        af[mi] = load_a16(sH1 + (mi * 16 + lrow) * SH + k0 + lhi * 8);
#pragma unroll
      for (int ni = 0; ni < 4; ++ni) {
        v16h bf = load_b16(W2t + (size_t)(nb + ni * 16 + lrow) * H_DIM + k0 + lhi * 8);
#pragma unroll
        for (int mi = 0; mi < 4; ++mi)
          c[mi][ni] = __builtin_amdgcn_wmma_f32_16x16x32_f16(
              false, af[mi], false, bf, (short)0, c[mi][ni], false, false);
      }
    }
#pragma unroll
    for (int mi = 0; mi < 4; ++mi)
#pragma unroll
      for (int ni = 0; ni < 4; ++ni)
#pragma unroll
        for (int j = 0; j < 8; ++j) {
          int r  = mi * 16 + lhi * 8 + j;
          int cc = nb + ni * 16 + lrow;
          float v = c[mi][ni][j] + b2[cc];
          v = v > 0.f ? v : 0.01f * v;
          sH2[r * SH + cc] = (_Float16)v;
        }
  }
  __syncthreads();

  // ---- layer 3: [64,512] @ [512,256(pad)] -> f32 logits in LDS ----
  {
    const int nb = wv * 32;
    v8f c[4][2] = {};
    for (int k0 = 0; k0 < H_DIM; k0 += 32) {
      v16h af[4];
#pragma unroll
      for (int mi = 0; mi < 4; ++mi)
        af[mi] = load_a16(sH2 + (mi * 16 + lrow) * SH + k0 + lhi * 8);
#pragma unroll
      for (int ni = 0; ni < 2; ++ni) {
        v16h bf = load_b16(W3t + (size_t)(nb + ni * 16 + lrow) * H_DIM + k0 + lhi * 8);
#pragma unroll
        for (int mi = 0; mi < 4; ++mi)
          c[mi][ni] = __builtin_amdgcn_wmma_f32_16x16x32_f16(
              false, af[mi], false, bf, (short)0, c[mi][ni], false, false);
      }
    }
#pragma unroll
    for (int mi = 0; mi < 4; ++mi)
#pragma unroll
      for (int ni = 0; ni < 2; ++ni)
#pragma unroll
        for (int j = 0; j < 8; ++j) {
          int r  = mi * 16 + lhi * 8 + j;
          int cc = nb + ni * 16 + lrow;
          sLog[r * SL + cc] = c[mi][ni][j] + (cc < NATOMS ? b3[cc] : 0.f);
        }
  }
  __syncthreads();

  // ---- zero proj accumulator (aliases dead sH2) ----
  for (int i = tid; i < MT * SP; i += 256) sProj[i] = 0.f;
  __syncthreads();

  // ---- softmax + Bellman projection, 8 rows per wave ----
  for (int rl = 0; rl < 8; ++rl) {
    int r = wv * 8 + rl;
    int grow = m0 + r;
    float lg[8];
#pragma unroll
    for (int i = 0; i < 8; ++i) {
      int a = lane + i * 32;
      lg[i] = (a < NATOMS) ? sLog[r * SL + a] : -3.0e38f;
    }
    float mx = lg[0];
#pragma unroll
    for (int i = 1; i < 8; ++i) mx = fmaxf(mx, lg[i]);
#pragma unroll
    for (int off = 16; off > 0; off >>= 1) mx = fmaxf(mx, __shfl_xor(mx, off, 32));
    float e[8]; float s = 0.f;
#pragma unroll
    for (int i = 0; i < 8; ++i) {
      int a = lane + i * 32;
      float ev = (a < NATOMS) ? __expf(lg[i] - mx) : 0.f;
      e[i] = ev; s += ev;
    }
#pragma unroll
    for (int off = 16; off > 0; off >>= 1) s += __shfl_xor(s, off, 32);
    float inv = 1.f / s;
    float rr = rew[grow];
    float bd = boot[grow] * disc[grow];
#pragma unroll
    for (int i = 0; i < 8; ++i) {
      int a = lane + i * 32;
      if (a < NATOMS) {
        float p  = e[i] * inv;
        float tz = rr + bd * qsup[a];
        tz = fminf(fmaxf(tz, -100.f), 100.f);
        float bfv = (tz + 100.f) * 1.25f;          // (tz - Vmin)/dz, dz = 0.8
        int l = (int)floorf(bfv);
        int u = (int)ceilf(bfv);
        bool eq = (l == u);
        int l2 = (eq && (u > 0))          ? l - 1 : l;
        int u2 = (eq && (l < NATOMS - 1)) ? u + 1 : u;
        atomicAdd(&sProj[r * SP + l2], p * ((float)u2 - bfv));
        atomicAdd(&sProj[r * SP + u2], p * (bfv - (float)l2));
      }
    }
  }
  __syncthreads();

  // ---- writeout ----
  for (int i = tid; i < MT * NATOMS; i += 256) {
    int r = i / NATOMS, c = i % NATOMS;
    out[(size_t)(m0 + r) * NATOMS + c] = sProj[r * SP + c];
  }
}

extern "C" void kernel_launch(void* const* d_in, const int* in_sizes, int n_in,
                              void* d_out, int out_size, void* d_ws, size_t ws_size,
                              hipStream_t stream) {
  const float* obs  = (const float*)d_in[0];
  const float* act  = (const float*)d_in[1];
  const float* rew  = (const float*)d_in[2];
  const float* boot = (const float*)d_in[3];
  const float* disc = (const float*)d_in[4];
  const float* qsup = (const float*)d_in[5];
  const float* W1   = (const float*)d_in[6];
  const float* b1   = (const float*)d_in[7];
  const float* W2   = (const float*)d_in[8];
  const float* b2   = (const float*)d_in[9];
  const float* W3   = (const float*)d_in[10];
  const float* b3   = (const float*)d_in[11];
  float* out = (float*)d_out;

  _Float16* W1t = (_Float16*)d_ws;                 // [512][128]
  _Float16* W2t = W1t + IN_DIM * H_DIM;            // [512][512]
  _Float16* W3t = W2t + H_DIM * H_DIM;             // [256][512]

  const int B = in_sizes[2];                       // rewards: [B]

  prep_weights<<<512, 256, 0, stream>>>(W1, W2, W3, W1t, W2t, W3t);
  fused_c51<<<B / MT, 256, SMEM_BYTES, stream>>>(obs, act, rew, boot, disc, qsup,
                                                 b1, b2, b3, W1t, W2t, W3t, out);
}
